// PLSTM_53781580480721
// MI455X (gfx1250) — compile-verified, matrix operand order checked by reading, not run on hardware
//
#include <hip/hip_runtime.h>
#include <hip/hip_bf16.h>
#include <math.h>

// Phased LSTM for MI455X (gfx1250, wave32, WMMA + async-to-LDS DMA).
//   K1: transpose/convert W [128,1024] and U [256,1024] to bf16 [N,K] layout.
//   K2: G = x@W + bias   (big parallel bf16 WMMA GEMM, [131072,128]x[128,1024])
//   K3: persistent recurrence, 8 WGs x 16 batch rows, h/c/gates in LDS,
//       h@U via v_wmma_f32_16x16x32_bf16, G tiles double-buffered into LDS
//       via global_load_async_to_lds_b128 (ASYNCcnt), 1024 serial steps.

typedef __bf16 bf16_t;
typedef __bf16 v16bf __attribute__((ext_vector_type(16)));
typedef float  v8f   __attribute__((ext_vector_type(8)));

#define B_DIM 128
#define S_DIM 1024
#define I_DIM 128
#define H_DIM 256
#define G_DIM 1024  // 4*H

// Dynamic LDS layout for K3 (bytes from base of the dynamic allocation):
#define L_GATES 0        // 16*1024*4 = 65536
#define L_CS    65536    // 16*256*4  = 16384
#define L_HF    81920    // 16*256*4  = 16384
#define L_HBF   98304    // 16*256*2  =  8192
#define L_PER   106496   // 256*4
#define L_SHF   107520
#define L_ONM   108544
#define L_ONE   109568
#define L_STAGE 110592   // 2 * 16*1024*4 = 131072
#define L_TOTAL 241664

union BF8 { uint4 u; bf16_t h[8]; };

// Load a 16-element bf16 fragment slice: 8 contiguous elems at p, 8 at p+16,
// matching the 16-bit A/B VGPR layout (K groups 0-7 / 16-23 per lane-half,
// lane-half K offset handled by the caller via +8).
__device__ __forceinline__ v16bf load_frag_bf16(const bf16_t* p) {
  BF8 lo, hi;
  lo.u = *(const uint4*)(p);
  hi.u = *(const uint4*)(p + 16);
  v16bf a;
#pragma unroll
  for (int e = 0; e < 8; ++e) { a[e] = lo.h[e]; a[e + 8] = hi.h[e]; }
  return a;
}

__device__ __forceinline__ v8f wmma_bf16(v16bf a, v16bf b, v8f c) {
  // (neg_a, A, neg_b, B, c_mod, C, reuse_a, reuse_b)
  return __builtin_amdgcn_wmma_f32_16x16x32_bf16(
      false, a, false, b, (short)0, c, false, false);
}

// Async DMA: 16 bytes global -> LDS, tracked by ASYNCcnt (gfx1250).
__device__ __forceinline__ void async_load_b128(unsigned lds_off, const void* gaddr) {
  asm volatile("global_load_async_to_lds_b128 %0, %1, off"
               :: "v"(lds_off), "v"(gaddr)
               : "memory");
}
__device__ __forceinline__ void wait_asynccnt0() {
  asm volatile("s_wait_asynccnt 0x0" ::: "memory");
}

// ---------------------------------------------------------------------------
// K1: weight transpose + f32 -> bf16 conversion.
//   Ut[n][k] = U[k][n]   (n<1024, k<256)
//   Wt[n][k] = W[k][n]   (n<1024, k<128)
// ---------------------------------------------------------------------------
__global__ void plstm_prep(const float* __restrict__ W, const float* __restrict__ U,
                           bf16_t* __restrict__ Wt, bf16_t* __restrict__ Ut) {
  int n = blockIdx.x;   // 0..1023
  int k = threadIdx.x;  // 0..255
  Ut[(size_t)n * H_DIM + k] = (bf16_t)U[(size_t)k * G_DIM + n];
  if (k < I_DIM) Wt[(size_t)n * I_DIM + k] = (bf16_t)W[(size_t)k * G_DIM + n];
}

// ---------------------------------------------------------------------------
// K2: G[m,n] = sum_k x[m,k]*W[k,n] + bias[n],  M = B*S = 131072, K=128, N=1024
// Block = 256 threads (8 waves). Wave w -> rows [blockY*128 + w*16, +16),
// cols [blockX*64, +64) as 4 WMMA tiles, K-loop of 4 x 32.
// ---------------------------------------------------------------------------
__global__ void plstm_gemm_x(const float* __restrict__ X,    // [M, 128] f32
                             const bf16_t* __restrict__ Wt,  // [1024, 128] bf16
                             const float* __restrict__ bias, // [1024]
                             float* __restrict__ G) {        // [M, 1024] f32
  const int lane = threadIdx.x & 31;
  const int wave = threadIdx.x >> 5;          // 0..7
  const int row0 = blockIdx.y * 128 + wave * 16;
  const int col0 = blockIdx.x * 64;
  const int r    = lane & 15;                 // A row / B col within tile
  const int kh   = (lane >> 4) << 3;          // lane-half K offset: 0 or 8
  const int rb   = kh;                        // D row offset: 0 or 8

  v8f acc[4] = {};

#pragma unroll
  for (int k0 = 0; k0 < I_DIM; k0 += 32) {
    // A fragment: x row slice, f32 -> bf16 on the fly.
    const float* xp = X + (size_t)(row0 + r) * I_DIM + k0 + kh;
    float4 f0 = *(const float4*)(xp + 0);
    float4 f1 = *(const float4*)(xp + 4);
    float4 f2 = *(const float4*)(xp + 16);
    float4 f3 = *(const float4*)(xp + 20);
    v16bf a;
    a[0]  = (bf16_t)f0.x; a[1]  = (bf16_t)f0.y; a[2]  = (bf16_t)f0.z; a[3]  = (bf16_t)f0.w;
    a[4]  = (bf16_t)f1.x; a[5]  = (bf16_t)f1.y; a[6]  = (bf16_t)f1.z; a[7]  = (bf16_t)f1.w;
    a[8]  = (bf16_t)f2.x; a[9]  = (bf16_t)f2.y; a[10] = (bf16_t)f2.z; a[11] = (bf16_t)f2.w;
    a[12] = (bf16_t)f3.x; a[13] = (bf16_t)f3.y; a[14] = (bf16_t)f3.z; a[15] = (bf16_t)f3.w;
#pragma unroll
    for (int nt = 0; nt < 4; ++nt) {
      int col = col0 + nt * 16 + r;
      v16bf b = load_frag_bf16(Wt + (size_t)col * I_DIM + k0 + kh);
      acc[nt] = wmma_bf16(a, b, acc[nt]);
    }
  }

#pragma unroll
  for (int nt = 0; nt < 4; ++nt) {
    int col = col0 + nt * 16 + r;
    float bv = bias[col];
#pragma unroll
    for (int rr = 0; rr < 8; ++rr) {
      int m = row0 + rb + rr;
      G[(size_t)m * G_DIM + col] = acc[nt][rr] + bv;
    }
  }
}

// ---------------------------------------------------------------------------
// K3: persistent recurrence. Grid = 8 blocks (16 batch rows each), 512 thr
// (16 waves). Wave w owns gate columns [w*64, +64). h kept in LDS as bf16
// (WMMA A operand) + f32 (exact blend state); c in LDS f32. Next step's
// precomputed x@W+bias tile is DMA'd into a double-buffered LDS stage with
// global_load_async_to_lds_b128, overlapped with the current step's work.
// ---------------------------------------------------------------------------
__global__ void plstm_recur(const float* __restrict__ G,    // [B*S, 1024] xW+bias
                            const bf16_t* __restrict__ Ut,  // [1024, 256] bf16
                            const float* __restrict__ ts,   // [B, S]
                            const float* __restrict__ Periods,
                            const float* __restrict__ Shifts,
                            const float* __restrict__ OnEnd,
                            float* __restrict__ out) {      // hseq | h_T | c_T
  extern __shared__ char smem[];
  float*  gates  = (float*)(smem + L_GATES);
  float*  c_s    = (float*)(smem + L_CS);
  float*  h_f    = (float*)(smem + L_HF);
  bf16_t* h_bf   = (bf16_t*)(smem + L_HBF);
  float*  per_s  = (float*)(smem + L_PER);
  float*  shf_s  = (float*)(smem + L_SHF);
  float*  onm_s  = (float*)(smem + L_ONM);
  float*  one_s  = (float*)(smem + L_ONE);
  float*  gstage = (float*)(smem + L_STAGE);  // 2 x [16][1024] f32

  const int tid  = threadIdx.x;   // 0..511
  const int lane = tid & 31;
  const int wave = tid >> 5;      // 0..15
  const int bt   = blockIdx.x;    // batch tile (16 rows)
  const int r    = lane & 15;
  const int kh   = (lane >> 4) << 3;  // 0 or 8
  const int rb   = kh;

  // init state + phase params
  for (int idx = tid; idx < 16 * H_DIM; idx += 512) {
    c_s[idx] = 0.0f; h_f[idx] = 0.0f; h_bf[idx] = (bf16_t)0.0f;
  }
  if (tid < H_DIM) {
    float p  = fabsf(Periods[tid]);
    float oe = fabsf(OnEnd[tid]) * p;
    per_s[tid] = p; one_s[tid] = oe; onm_s[tid] = 0.5f * oe;
    shf_s[tid] = Shifts[tid];
  }
  __syncthreads();

  const int    col0 = wave * 64;
  const size_t grow = (size_t)bt * 16;  // first global batch row of this tile

  for (int t = 0; t < S_DIM; ++t) {
    // ---- kick off async DMA of step t+1's G tile into LDS stage ----
    if (t + 1 < S_DIM) {
      const unsigned buf = (unsigned)((t + 1) & 1);
#pragma unroll
      for (int i = 0; i < 8; ++i) {
        int      l       = tid + i * 512;      // 0..4095 (16B chunks)
        int      row     = l >> 8;             // 0..15   (256 chunks per row)
        unsigned off     = (unsigned)(l & 255) << 4;  // byte offset in 4KB row
        unsigned lds_off = (unsigned)L_STAGE + buf * 65536u +
                           (unsigned)row * 4096u + off;
        const char* gp = (const char*)G +
            (((grow + (size_t)row) * S_DIM + (size_t)(t + 1)) * G_DIM) *
                sizeof(float) + off;
        async_load_b128(lds_off, gp);
      }
    }

    // ---- recurrent GEMM: this wave's 64 gate columns of h @ U ----
    v8f acc[4] = {};
#pragma unroll
    for (int k0 = 0; k0 < H_DIM; k0 += 32) {
      v16bf a = load_frag_bf16(h_bf + r * H_DIM + k0 + kh);  // LDS b128 reads
#pragma unroll
      for (int nt = 0; nt < 4; ++nt) {
        int col = col0 + nt * 16 + r;
        v16bf b = load_frag_bf16(Ut + (size_t)col * H_DIM + k0 + kh);
        acc[nt] = wmma_bf16(a, b, acc[nt]);
      }
    }

    // ---- write gates to LDS, fusing precomputed x@W + bias ----
    const float* gsrc = gstage + (size_t)(t & 1) * (16 * G_DIM);
#pragma unroll
    for (int nt = 0; nt < 4; ++nt) {
      int col = col0 + nt * 16 + r;
#pragma unroll
      for (int rr = 0; rr < 8; ++rr) {
        int lr = rb + rr;
        float pre = (t == 0)
            ? G[(grow + (size_t)lr) * S_DIM * G_DIM + col]  // t=0: direct
            : gsrc[lr * G_DIM + col];                       // staged by DMA
        gates[lr * G_DIM + col] = acc[nt][rr] + pre;
      }
    }
    __syncthreads();

    // ---- elementwise PLSTM cell + phase gate: 16 x 256 ----
#pragma unroll
    for (int k = 0; k < 8; ++k) {
      int idx = tid + k * 512;
      int row = idx >> 8;    // 0..15
      int j   = idx & 255;   // hidden unit
      float ig = gates[row * G_DIM + j];
      float fg = gates[row * G_DIM + 256 + j];
      float gg = gates[row * G_DIM + 512 + j];
      float og = gates[row * G_DIM + 768 + j];
      float it = 1.0f / (1.0f + __expf(-ig));
      float ft = 1.0f / (1.0f + __expf(-fg));
      float gt = tanhf(gg);
      float ot = 1.0f / (1.0f + __expf(-og));
      float c_old = c_s[row * H_DIM + j];
      float h_old = h_f[row * H_DIM + j];
      float cn = ft * c_old + it * gt;
      float hn = ot * tanhf(cn);
      int   b  = bt * 16 + row;
      float tt = ts[(size_t)b * S_DIM + t];
      float per = per_s[j];
      float ic  = fmodf(tt + shf_s[j], per);
      float onm = onm_s[j], oe = one_s[j];
      float mask = (ic <= onm) ? (ic / onm)
                 : ((ic <= oe) ? ((oe - ic) / onm)
                               : (0.001f * (ic / per)));
      float c2 = mask * cn + (1.0f - mask) * c_old;
      float h2 = mask * hn + (1.0f - mask) * h_old;
      c_s[row * H_DIM + j] = c2;
      h_f[row * H_DIM + j] = h2;
      h_bf[row * H_DIM + j] = (bf16_t)h2;
      out[(size_t)b * S_DIM * H_DIM + (size_t)t * H_DIM + j] = h2;
    }

    // Ensure this wave's DMA (issued at the top of this step, consumed next
    // step) has fully landed in LDS, then make it visible to all waves.
    wait_asynccnt0();
    __syncthreads();
  }

  // finals: h_T, c_T
  const size_t hseq = (size_t)B_DIM * S_DIM * H_DIM;
#pragma unroll
  for (int k = 0; k < 8; ++k) {
    int idx = tid + k * 512;
    int row = idx >> 8, j = idx & 255;
    int b = bt * 16 + row;
    out[hseq + (size_t)b * H_DIM + j] = h_f[row * H_DIM + j];
    out[hseq + (size_t)B_DIM * H_DIM + (size_t)b * H_DIM + j] = c_s[row * H_DIM + j];
  }
}

// ---------------------------------------------------------------------------
extern "C" void kernel_launch(void* const* d_in, const int* in_sizes, int n_in,
                              void* d_out, int out_size, void* d_ws, size_t ws_size,
                              hipStream_t stream) {
  (void)in_sizes; (void)n_in; (void)out_size; (void)ws_size;
  const float* x    = (const float*)d_in[0];  // [128,1024,128]
  const float* ts   = (const float*)d_in[1];  // [128,1024]
  const float* W    = (const float*)d_in[2];  // [128,1024]
  const float* U    = (const float*)d_in[3];  // [256,1024]
  const float* bias = (const float*)d_in[4];  // [1024]
  const float* Per  = (const float*)d_in[5];  // [256]
  const float* Shf  = (const float*)d_in[6];  // [256]
  const float* OnE  = (const float*)d_in[7];  // [256]
  float* out = (float*)d_out;

  char* ws = (char*)d_ws;
  const size_t Gbytes = (size_t)B_DIM * S_DIM * G_DIM * sizeof(float);  // 512 MiB
  float*  Gp = (float*)ws;
  bf16_t* Ut = (bf16_t*)(ws + Gbytes);
  bf16_t* Wt = (bf16_t*)(ws + Gbytes + (size_t)G_DIM * H_DIM * sizeof(bf16_t));

  plstm_prep<<<dim3(G_DIM), dim3(H_DIM), 0, stream>>>(W, U, Wt, Ut);
  plstm_gemm_x<<<dim3(G_DIM / 64, (B_DIM * S_DIM) / 128), dim3(256), 0, stream>>>(
      x, Wt, bias, Gp);
  plstm_recur<<<dim3(B_DIM / 16), dim3(512), L_TOTAL, stream>>>(
      Gp, Ut, ts, Per, Shf, OnE, out);
}
